// GraphSAGE_50474455663065
// MI455X (gfx1250) — compile-verified, matrix operand order checked by reading, not run on hardware
//
#include <hip/hip_runtime.h>

typedef __attribute__((ext_vector_type(2)))  float  v2f;
typedef __attribute__((ext_vector_type(8)))  float  v8f;
typedef __attribute__((ext_vector_type(16))) __bf16 v16bf;

#define N_NODES 40000
#define FEAT    128

// ---------------------------------------------------------------------------
// degree / inverse-degree
// ---------------------------------------------------------------------------
__global__ void deg_kernel(const int* __restrict__ dst, float* __restrict__ deg, int nE) {
    int e = blockIdx.x * blockDim.x + threadIdx.x;
    if (e < nE) atomicAdd(&deg[dst[e]], 1.0f);
}

__global__ void invdeg_kernel(float* __restrict__ deg, int n) {
    int i = blockIdx.x * blockDim.x + threadIdx.x;
    if (i < n) deg[i] = 1.0f / fmaxf(deg[i], 1.0f);
}

// ---------------------------------------------------------------------------
// edge aggregation: one wave per edge, one float4 chunk per lane (coalesced)
// ---------------------------------------------------------------------------
__global__ void aggregate_kernel(const float* __restrict__ H,
                                 const int* __restrict__ src,
                                 const int* __restrict__ dst,
                                 float* __restrict__ agg, int nE) {
    long t = (long)blockIdx.x * blockDim.x + threadIdx.x;
    long e = t >> 5;               // 32 lanes cover 128 floats of one edge
    int  c = (int)(t & 31);
    if (e >= nE) return;
    int s = src[e], d = dst[e];
    const float4 v = ((const float4*)(H + (long)s * FEAT))[c];
    float* o = agg + (long)d * FEAT + (long)c * 4;
    atomicAdd(o + 0, v.x);
    atomicAdd(o + 1, v.y);
    atomicAdd(o + 2, v.z);
    atomicAdd(o + 3, v.w);
}

// ---------------------------------------------------------------------------
// fused SAGE GEMM:  Out = act( Hin@Wself + (Agg*invdeg)@Wneigh + b )
// block = (NOUT/16) waves; wave w computes the 16x16 tile at columns [16w,16w+16)
// for the 16-row M tile owned by this block. K = FEAT = 128.
// ---------------------------------------------------------------------------
template <int NOUT, bool RELU>
__global__ __launch_bounds__(NOUT * 2)
void sage_gemm_kernel(const float* __restrict__ Hin,
                      const float* __restrict__ Agg,
                      const float* __restrict__ invdeg,
                      const float* __restrict__ Wself,
                      const float* __restrict__ Wneigh,
                      const float* __restrict__ bias,
                      float* __restrict__ Out) {
    __shared__ float As[16][FEAT + 4];   // self features   (pad 4 -> conflict free)
    __shared__ float An[16][FEAT + 4];   // mean neighbor features

    const int tid = threadIdx.x;
    const int m0  = blockIdx.x * 16;

    // stage the 16x128 A tiles (self + scaled neighbor) into LDS
    for (int i = tid; i < 16 * FEAT; i += NOUT * 2) {
        int r = i >> 7, c = i & (FEAT - 1);
        long g = (long)(m0 + r) * FEAT + c;
        As[r][c] = Hin[g];
        An[r][c] = Agg[g] * invdeg[m0 + r];
    }
    __syncthreads();

    const int wave  = tid >> 5;
    const int lane  = tid & 31;
    const int n0    = wave * 16;
    const int mrow  = lane & 15;        // A-frag row / B,C column index
    const int half  = lane >> 4;        // lane group selects K sub-block

    v8f acc = {};

#if __has_builtin(__builtin_amdgcn_wmma_f32_16x16x4_f32)
    // native f32 WMMA path: A 16x4 frag = 2 f32/lane (K = half*2 + {0,1})
    const int kh = half * 2;
#pragma unroll
    for (int k = 0; k < FEAT; k += 4) {
        v2f a, b;
        a.x = As[mrow][k + kh];
        a.y = As[mrow][k + kh + 1];
        b.x = Wself[(long)(k + kh)     * NOUT + n0 + mrow];
        b.y = Wself[(long)(k + kh + 1) * NOUT + n0 + mrow];
        acc = __builtin_amdgcn_wmma_f32_16x16x4_f32(false, a, false, b,
                                                    (short)0, acc, false, false);
    }
#pragma unroll
    for (int k = 0; k < FEAT; k += 4) {
        v2f a, b;
        a.x = An[mrow][k + kh];
        a.y = An[mrow][k + kh + 1];
        b.x = Wneigh[(long)(k + kh)     * NOUT + n0 + mrow];
        b.y = Wneigh[(long)(k + kh + 1) * NOUT + n0 + mrow];
        acc = __builtin_amdgcn_wmma_f32_16x16x4_f32(false, a, false, b,
                                                    (short)0, acc, false, false);
    }
#else
    // fallback: bf16 WMMA (codegen-confirmed). 16-bit A 16x32 layout:
    // lane group 'half': j=0..7 -> K = 8*half + j ; j=8..15 -> K = 16 + 8*half + (j-8)
    const int kb = half * 8;
#pragma unroll
    for (int k = 0; k < FEAT; k += 32) {
        v16bf a, b;
#pragma unroll
        for (int j = 0; j < 8; ++j) {
            int k1 = k + kb + j, k2 = k + 16 + kb + j;
            a[j]     = (__bf16)As[mrow][k1];
            a[j + 8] = (__bf16)As[mrow][k2];
            b[j]     = (__bf16)Wself[(long)k1 * NOUT + n0 + mrow];
            b[j + 8] = (__bf16)Wself[(long)k2 * NOUT + n0 + mrow];
        }
        acc = __builtin_amdgcn_wmma_f32_16x16x32_bf16(false, a, false, b,
                                                      (short)0, acc, false, false);
    }
#pragma unroll
    for (int k = 0; k < FEAT; k += 32) {
        v16bf a, b;
#pragma unroll
        for (int j = 0; j < 8; ++j) {
            int k1 = k + kb + j, k2 = k + 16 + kb + j;
            a[j]     = (__bf16)An[mrow][k1];
            a[j + 8] = (__bf16)An[mrow][k2];
            b[j]     = (__bf16)Wneigh[(long)k1 * NOUT + n0 + mrow];
            b[j + 8] = (__bf16)Wneigh[(long)k2 * NOUT + n0 + mrow];
        }
        acc = __builtin_amdgcn_wmma_f32_16x16x32_bf16(false, a, false, b,
                                                      (short)0, acc, false, false);
    }
#endif

    // C/D layout: VGPR i -> M = i + 8*half, N = lane&15
    const float bv    = bias[n0 + mrow];
    const int   mbase = m0 + half * 8;
#pragma unroll
    for (int i = 0; i < 8; ++i) {
        float v = acc[i] + bv;
        if (RELU) v = fmaxf(v, 0.0f);
        Out[(long)(mbase + i) * NOUT + n0 + mrow] = v;
    }
}

// ---------------------------------------------------------------------------
extern "C" void kernel_launch(void* const* d_in, const int* in_sizes, int n_in,
                              void* d_out, int out_size, void* d_ws, size_t ws_size,
                              hipStream_t stream) {
    (void)n_in; (void)out_size; (void)ws_size;

    const float* x       = (const float*)d_in[0];
    const int*   src     = (const int*)  d_in[1];
    const int*   dst     = (const int*)  d_in[2];
    const float* Wself0  = (const float*)d_in[3];
    const float* Wneigh0 = (const float*)d_in[4];
    const float* b0      = (const float*)d_in[5];
    const float* Wself1  = (const float*)d_in[6];
    const float* Wneigh1 = (const float*)d_in[7];
    const float* b1      = (const float*)d_in[8];
    const float* Wself2  = (const float*)d_in[9];
    const float* Wneigh2 = (const float*)d_in[10];
    const float* b2      = (const float*)d_in[11];
    const int nE = in_sizes[1];

    // workspace: deg (aligned) + agg[N,128]
    float* deg = (float*)d_ws;
    float* agg = deg + ((N_NODES + 255) & ~255);

    // output layout: (h3, h0, h1, h2, h3) flat
    float* h3a = (float*)d_out;
    float* h0  = h3a + (size_t)N_NODES * 64;
    float* h1  = h0  + (size_t)N_NODES * 128;
    float* h2  = h1  + (size_t)N_NODES * 128;
    float* h3b = h2  + (size_t)N_NODES * 128;

    const size_t featBytes = (size_t)N_NODES * FEAT * sizeof(float);
    const int aggBlocks = (int)(((long)nE * 32 + 255) / 256);

    // degrees -> inverse degrees (deterministic per call; scratch re-zeroed)
    hipMemsetAsync(deg, 0, N_NODES * sizeof(float), stream);
    deg_kernel<<<(nE + 255) / 256, 256, 0, stream>>>(dst, deg, nE);
    invdeg_kernel<<<(N_NODES + 255) / 256, 256, 0, stream>>>(deg, N_NODES);

    // h0 = x
    hipMemcpyAsync(h0, x, featBytes, hipMemcpyDeviceToDevice, stream);

    // layer 0: h1 = relu(x@Ws0 + mean@Wn0 + b0)
    hipMemsetAsync(agg, 0, featBytes, stream);
    aggregate_kernel<<<aggBlocks, 256, 0, stream>>>(x, src, dst, agg, nE);
    sage_gemm_kernel<128, true><<<N_NODES / 16, 256, 0, stream>>>(
        x, agg, deg, Wself0, Wneigh0, b0, h1);

    // layer 1: h2 = relu(h1@Ws1 + mean@Wn1 + b1)
    hipMemsetAsync(agg, 0, featBytes, stream);
    aggregate_kernel<<<aggBlocks, 256, 0, stream>>>(h1, src, dst, agg, nE);
    sage_gemm_kernel<128, true><<<N_NODES / 16, 256, 0, stream>>>(
        h1, agg, deg, Wself1, Wneigh1, b1, h2);

    // layer 2: h3 = h2@Ws2 + mean@Wn2 + b2  (no relu, NOUT=64)
    hipMemsetAsync(agg, 0, featBytes, stream);
    aggregate_kernel<<<aggBlocks, 256, 0, stream>>>(h2, src, dst, agg, nE);
    sage_gemm_kernel<64, false><<<N_NODES / 16, 128, 0, stream>>>(
        h2, agg, deg, Wself2, Wneigh2, b2, h3a);

    // duplicate h3 into its second output slot
    hipMemcpyAsync(h3b, h3a, (size_t)N_NODES * 64 * sizeof(float),
                   hipMemcpyDeviceToDevice, stream);
}
